// TorchModel_3539053052209
// MI455X (gfx1250) — matrix operand order repaired and checked by reading, not compiled, not run on hardware
//
#include <hip/hip_runtime.h>
#include <hip/hip_bf16.h>

#define B_    4096
#define S_    512
#define E_    64
#define H_    64
#define NCLS_ 5
#define BN_EPS 1e-5f

typedef __attribute__((ext_vector_type(16))) __bf16 v16bf;
typedef __attribute__((ext_vector_type(8)))  __bf16 v8bf;
typedef __attribute__((ext_vector_type(8)))  float  v8f;

#if defined(__has_builtin)
#if __has_builtin(__builtin_amdgcn_tanhf)
#define TANHF(x) __builtin_amdgcn_tanhf(x)
#endif
#endif
#ifndef TANHF
#define TANHF(x) tanhf(x)
#endif

// ---------------------------------------------------------------------------
// Workspace layout (bytes):
//   0      : Wih_bf  [64*64] bf16   (8 KB)   B-operand source, row-major [n][k]
//   8192   : Whh_bf  [64*64] bf16   (8 KB)
//   16384  : bias    [64]  f32      (b_ih + b_hh, folded once)
//   16640  : mean    [64]  f32
//   16896  : rstd    [64]  f32
//   32768  : hbuf    [4096*64] f32  (1 MB)   h_n for BN + head
// ---------------------------------------------------------------------------

__global__ void prep_kernel(const float* __restrict__ Wih, const float* __restrict__ Whh,
                            const float* __restrict__ bih, const float* __restrict__ bhh,
                            __bf16* __restrict__ Wih_bf, __bf16* __restrict__ Whh_bf,
                            float* __restrict__ bias) {
  int i = blockIdx.x * blockDim.x + threadIdx.x;
  if (i < H_ * E_) {
    Wih_bf[i] = (__bf16)Wih[i];
    Whh_bf[i] = (__bf16)Whh[i];
  }
  if (i < H_) bias[i] = bih[i] + bhh[i];
}

// One wave32 per 16-sequence tile. Weights live in VGPRs as WMMA B fragments.
// Per timestep: 16x v_wmma_f32_16x16x32_bf16 (two [16,64]x[64,64] GEMMs).
// Embedding gather is software-pipelined (tokens depth-2, emb rows depth-1)
// so the only true serial dependency per step is h -> WMMA -> tanh -> h.
__launch_bounds__(128, 1)
__global__ void rnn_kernel(const int* __restrict__ x, const float* __restrict__ emb,
                           const __bf16* __restrict__ Wih_bf, const __bf16* __restrict__ Whh_bf,
                           const float* __restrict__ bias, float* __restrict__ hbuf) {
  // per-wave private h staging: 16 rows x 72 bf16 (pad 8 breaks 128B bank stride)
  __shared__ __bf16 hlds[4][16][72];

  const int lane = threadIdx.x & 31;
  const int w    = threadIdx.x >> 5;
  const int tile = blockIdx.x * 4 + w;
  const int row0 = tile * 16;
  const int lh   = lane & 15;   // 0..15
  const int hi   = lane >> 4;   // 0 or 1

  // --- Load B fragments (B[k][n] = W[n][k], i.e. W transposed on the fly).
  // bf16 B 32x16 layout: lane n<16 -> K {kb..kb+7, kb+16..kb+23}, lane n+16 -> same + 8.
  // => each lane reads two 8-bf16 (16B) chunks from row (nt*16+lh), cols (kt*32+hi*8) and +16.
  v16bf bih_f[2][4], bhh_f[2][4];
#pragma unroll
  for (int kt = 0; kt < 2; ++kt) {
#pragma unroll
    for (int nt = 0; nt < 4; ++nt) {
      const int r = nt * 16 + lh;
      const int k = kt * 32 + hi * 8;
      const v8bf i0 = *(const v8bf*)(Wih_bf + r * E_ + k), i1 = *(const v8bf*)(Wih_bf + r * E_ + k + 16);
      const v8bf h0 = *(const v8bf*)(Whh_bf + r * H_ + k), h1 = *(const v8bf*)(Whh_bf + r * H_ + k + 16);
      bih_f[kt][nt] = __builtin_shufflevector(i0, i1, 0,1,2,3,4,5,6,7,8,9,10,11,12,13,14,15);
      bhh_f[kt][nt] = __builtin_shufflevector(h0, h1, 0,1,2,3,4,5,6,7,8,9,10,11,12,13,14,15);
    }
  }

  // bias for accumulator init: D tile column n = nt*16 + (lane&15)
  float biasv[4];
#pragma unroll
  for (int nt = 0; nt < 4; ++nt) biasv[nt] = bias[nt * 16 + lh];

  // h(0) = 0
  for (int i = lane; i < 16 * 72; i += 32) (&hlds[w][0][0])[i] = (__bf16)0.0f;

  const int  arow     = row0 + lh;        // sequence this lane supplies as A row
  const int  hrow_off = hi * 8;           // D layout: lanes>=16 hold rows M=8..15
  const int* xrow     = x + (long)arow * S_;

  // ---- software-pipeline prologue -----------------------------------------
  // bf16 A 16x32 layout: lane m<16 -> K {kb..kb+7, kb+16..kb+23} (kb=kt*32),
  // lane m+16 -> same + 8.
  v16bf axe[2];
  {
    const int tok0 = xrow[0];
    const float* er = emb + (long)tok0 * E_;
#pragma unroll
    for (int kt = 0; kt < 2; ++kt) {
      const int kb = kt * 32 + hi * 8;
#pragma unroll
      for (int j = 0; j < 8; ++j) {
        axe[kt][j]     = (__bf16)er[kb + j];
        axe[kt][j + 8] = (__bf16)er[kb + 16 + j];
      }
    }
  }
  int tok_n1 = xrow[1 < S_ ? 1 : 0];      // token for t+1, already in flight

  v8f acc[4];
  for (int t = 0; t < S_; ++t) {
    // (1) issue token load for t+2 (a full iteration of latency slack)
    const int tok_n2 = xrow[(t + 2 < S_) ? t + 2 : t];

    // (2) issue raw-f32 gather of t+1's embedding row; waits land after WMMA
    const float* ern = emb + (long)tok_n1 * E_;
    float nf[2][16];
#pragma unroll
    for (int kt = 0; kt < 2; ++kt) {
      const int kb = kt * 32 + hi * 8;
#pragma unroll
      for (int j = 0; j < 8; ++j) {
        nf[kt][j]     = ern[kb + j];
        nf[kt][j + 8] = ern[kb + 16 + j];
      }
    }

    // (3) acc = bias (broadcast per column)
#pragma unroll
    for (int nt = 0; nt < 4; ++nt) {
#pragma unroll
      for (int r = 0; r < 8; ++r) acc[nt][r] = biasv[nt];
    }

    // (4) WMMA block: acc += xe@Wih^T + h@Whh^T
#pragma unroll
    for (int kt = 0; kt < 2; ++kt) {
      const int kbase = kt * 32 + hi * 8;
      union { v16bf v; v8bf h[2]; } ah;   // h A fragment from LDS staging
      ah.h[0] = *(const v8bf*)&hlds[w][lh][kbase];
      ah.h[1] = *(const v8bf*)&hlds[w][lh][kbase + 16];
#pragma unroll
      for (int nt = 0; nt < 4; ++nt) {
        acc[nt] = __builtin_amdgcn_wmma_f32_16x16x32_bf16(
            false, axe[kt], false, bih_f[kt][nt], (short)0, acc[nt], false, false);
        acc[nt] = __builtin_amdgcn_wmma_f32_16x16x32_bf16(
            false, ah.v,    false, bhh_f[kt][nt], (short)0, acc[nt], false, false);
      }
    }

    // (5) h = tanh(acc); keep f32 in acc, stage bf16 copy for next step's A
#pragma unroll
    for (int nt = 0; nt < 4; ++nt) {
#pragma unroll
      for (int r = 0; r < 8; ++r) {
        const float hv = TANHF(acc[nt][r]);
        acc[nt][r] = hv;
        hlds[w][hrow_off + r][nt * 16 + lh] = (__bf16)hv;
      }
    }

    // (6) rotate pipeline: convert prefetched row into next A fragment
#pragma unroll
    for (int kt = 0; kt < 2; ++kt) {
#pragma unroll
      for (int j = 0; j < 16; ++j) axe[kt][j] = (__bf16)nf[kt][j];
    }
    tok_n1 = tok_n2;
  }

  // h_n (f32) -> workspace for BN + head
#pragma unroll
  for (int nt = 0; nt < 4; ++nt) {
#pragma unroll
    for (int r = 0; r < 8; ++r) {
      hbuf[(long)(row0 + hrow_off + r) * H_ + nt * 16 + lh] = acc[nt][r];
    }
  }
}

// Deterministic per-column BN stats (tree reduction; no float atomics so replays
// are bit-stable).
__global__ void bnstats_kernel(const float* __restrict__ hbuf,
                               float* __restrict__ mean, float* __restrict__ rstd) {
  __shared__ float s1[256], s2[256];
  const int j = blockIdx.x;  // hidden column 0..63
  float sum = 0.f, sq = 0.f;
  for (int r = threadIdx.x; r < B_; r += 256) {
    const float v = hbuf[(long)r * H_ + j];
    sum += v;
    sq += v * v;
  }
  s1[threadIdx.x] = sum;
  s2[threadIdx.x] = sq;
  __syncthreads();
  for (int off = 128; off > 0; off >>= 1) {
    if (threadIdx.x < off) {
      s1[threadIdx.x] += s1[threadIdx.x + off];
      s2[threadIdx.x] += s2[threadIdx.x + off];
    }
    __syncthreads();
  }
  if (threadIdx.x == 0) {
    const float m   = s1[0] * (1.0f / B_);
    const float var = s2[0] * (1.0f / B_) - m * m;  // biased variance
    mean[j] = m;
    rstd[j] = rsqrtf(var + BN_EPS);
  }
}

// Fused BN-normalize + [64]x[64,5] head per row.
__global__ void head_kernel(const float* __restrict__ hbuf,
                            const float* __restrict__ mean, const float* __restrict__ rstd,
                            const float* __restrict__ gamma, const float* __restrict__ beta,
                            const float* __restrict__ Wfc, const float* __restrict__ bfc,
                            float* __restrict__ out) {
  const int r = blockIdx.x * blockDim.x + threadIdx.x;
  if (r >= B_) return;
  float dot[NCLS_];
#pragma unroll
  for (int c = 0; c < NCLS_; ++c) dot[c] = bfc[c];
  for (int k = 0; k < H_; ++k) {
    const float hn = (hbuf[(long)r * H_ + k] - mean[k]) * rstd[k] * gamma[k] + beta[k];
#pragma unroll
    for (int c = 0; c < NCLS_; ++c) dot[c] += hn * Wfc[c * H_ + k];
  }
#pragma unroll
  for (int c = 0; c < NCLS_; ++c) out[r * NCLS_ + c] = dot[c];
}

extern "C" void kernel_launch(void* const* d_in, const int* in_sizes, int n_in,
                              void* d_out, int out_size, void* d_ws, size_t ws_size,
                              hipStream_t stream) {
  const int*   x     = (const int*)  d_in[0];
  const float* emb   = (const float*)d_in[1];
  const float* W_ih  = (const float*)d_in[2];
  const float* W_hh  = (const float*)d_in[3];
  const float* b_ih  = (const float*)d_in[4];
  const float* b_hh  = (const float*)d_in[5];
  const float* gamma = (const float*)d_in[6];
  const float* beta  = (const float*)d_in[7];
  const float* W_fc  = (const float*)d_in[8];
  const float* b_fc  = (const float*)d_in[9];
  float* out = (float*)d_out;

  char* ws = (char*)d_ws;
  __bf16* Wih_bf = (__bf16*)(ws + 0);
  __bf16* Whh_bf = (__bf16*)(ws + 8192);
  float*  bias   = (float*) (ws + 16384);
  float*  mean   = (float*) (ws + 16640);
  float*  rstd   = (float*) (ws + 16896);
  float*  hbuf   = (float*) (ws + 32768);

  prep_kernel<<<(H_ * E_ + 255) / 256, 256, 0, stream>>>(W_ih, W_hh, b_ih, b_hh,
                                                         Wih_bf, Whh_bf, bias);
  rnn_kernel<<<B_ / 64, 128, 0, stream>>>(x, emb, Wih_bf, Whh_bf, bias, hbuf);
  bnstats_kernel<<<H_, 256, 0, stream>>>(hbuf, mean, rstd);
  head_kernel<<<(B_ + 255) / 256, 256, 0, stream>>>(hbuf, mean, rstd, gamma, beta,
                                                    W_fc, b_fc, out);
}
